// CausalSelfAttention2D_67989332296382
// MI455X (gfx1250) — compile-verified, matrix operand order checked by reading, not run on hardware
//
#include <hip/hip_runtime.h>
#include <math.h>

// ---------------------------------------------------------------------------
// CDNA5 (gfx1250) causal 2D self-attention: bf16 WMMA + async global->LDS
// double-buffered pipelines (ASYNCcnt), vectorized ds_load_b128 fragments.
//   B=8, C=512, H=W=32 (L=1024), HEADS=8, D_HEAD=64
// Layouts (all chosen so fragments are 2x ds_load_b128 and staging is 16B/lane):
//   xn : [b][l][c]      q,k: [b][h][l][d]      v: [b][h][d][l]      ao: [b][l][c]
// ---------------------------------------------------------------------------

typedef __bf16 v16bf __attribute__((ext_vector_type(16)));
typedef __bf16 v8bf  __attribute__((ext_vector_type(8)));
typedef float  v8f   __attribute__((ext_vector_type(8)));

#define BATCH   8
#define CCH     512
#define LSEQ    1024
#define DMODEL  512
#define NHEADS  8
#define DHEAD   64

// ---- CDNA5 async global->LDS (tracked by ASYNCcnt) -------------------------
__device__ __forceinline__ void async_copy_b128(const void* lds_dst,
                                                const void* gsrc) {
  const unsigned lds = (unsigned)(size_t)lds_dst;      // LDS byte address
  const unsigned long long ga = (unsigned long long)(size_t)gsrc;
  asm volatile("global_load_async_to_lds_b128 %0, %1, off"
               :: "v"(lds), "v"(ga) : "memory");
}
#define ASYNC_WAIT_0 asm volatile("s_wait_asynccnt 0x0" ::: "memory")
#define ASYNC_WAIT_1 asm volatile("s_wait_asynccnt 0x1" ::: "memory")
#define ASYNC_WAIT_8 asm volatile("s_wait_asynccnt 0x8" ::: "memory")

__device__ __forceinline__ v8f wmma_bf16(v16bf a, v16bf b, v8f c) {
  return __builtin_amdgcn_wmma_f32_16x16x32_bf16(false, a, false, b, (short)0, c,
                                                 false, false);
}

// 16-bit operand fragment (ISA 7.12.2): two contiguous 16B chunks per lane.
__device__ __forceinline__ v16bf load_frag(const __bf16* rowbase, int hlf) {
  const v8bf lo = *(const v8bf*)(rowbase + hlf * 8);
  const v8bf hi = *(const v8bf*)(rowbase + 16 + hlf * 8);
  return __builtin_shufflevector(lo, hi, 0, 1, 2, 3, 4, 5, 6, 7,
                                 8, 9, 10, 11, 12, 13, 14, 15);
}

__device__ __forceinline__ v8bf cvt8(const float4 f0, const float4 f1) {
  v8bf r;
  r[0] = (__bf16)f0.x; r[1] = (__bf16)f0.y; r[2] = (__bf16)f0.z; r[3] = (__bf16)f0.w;
  r[4] = (__bf16)f1.x; r[5] = (__bf16)f1.y; r[6] = (__bf16)f1.z; r[7] = (__bf16)f1.w;
  return r;
}

// ---------------------------------------------------------------------------
// Kernel 1: channel layernorm, one wave32 per pixel; xn out as [b][l][c].
// ---------------------------------------------------------------------------
__global__ __launch_bounds__(256) void ln_kernel(const float* __restrict__ x,
                                                 const float* __restrict__ gam,
                                                 const float* __restrict__ bet,
                                                 __bf16* __restrict__ xn) {
  const int wv   = threadIdx.x >> 5;
  const int lane = threadIdx.x & 31;
  const int pix  = blockIdx.x * 8 + wv;
  const int b    = pix >> 10;
  const int l    = pix & 1023;
  const float* xb = x + (size_t)b * CCH * LSEQ + l;

  float vals[16];
  float s = 0.f, s2 = 0.f;
#pragma unroll
  for (int i = 0; i < 16; ++i) {
    const int c = lane + 32 * i;
    const float v = xb[(size_t)c * LSEQ];
    vals[i] = v;
    s += v;
    s2 += v * v;
  }
#pragma unroll
  for (int off = 16; off >= 1; off >>= 1) {
    s  += __shfl_xor(s, off, 32);
    s2 += __shfl_xor(s2, off, 32);
  }
  const float mean = s * (1.f / CCH);
  const float var  = s2 * (1.f / CCH) - mean * mean;
  const float inv  = rsqrtf(var + 1e-5f);

  __bf16* ob = xn + ((size_t)b * LSEQ + l) * CCH;
#pragma unroll
  for (int i = 0; i < 16; ++i) {
    const int c = lane + 32 * i;
    ob[c] = (__bf16)((vals[i] - mean) * inv * gam[c] + bet[c]);
  }
}

// ---------------------------------------------------------------------------
// Kernel 2: QKV GEMM (1536x512 @ 512xL per batch). B tile async double-buffered.
// 256 threads / 8 waves; tile 128(M) x 64(N); 16 K-steps of 32.
// ---------------------------------------------------------------------------
__global__ __launch_bounds__(256) void qkv_kernel(const __bf16* __restrict__ xn,
                                                  const float* __restrict__ w,
                                                  const float* __restrict__ bias,
                                                  __bf16* __restrict__ qo,
                                                  __bf16* __restrict__ ko,
                                                  __bf16* __restrict__ vo) {
  __shared__ __bf16 As[128][40];       // [m][k] stride 80B
  __shared__ __bf16 Bs[2][64][40];     // double-buffered [n][k]

  const int b    = blockIdx.z;
  const int m0   = blockIdx.y * 128;
  const int n0   = blockIdx.x * 64;
  const int tid  = threadIdx.x;
  const int lane = tid & 31;
  const int wv   = tid >> 5;
  const int hlf  = lane >> 4;
  const int nn   = lane & 15;

  const __bf16* xb = xn + (size_t)b * LSEQ * CCH;

  // per-lane staging slot for the async B pipeline
  const int bn = tid >> 2, bc = (tid & 3) * 8;
  const __bf16* gsrc = xb + (size_t)(n0 + bn) * CCH + bc;
  void* ldsB[2] = { &Bs[0][bn][bc], &Bs[1][bn][bc] };

  v8f acc[4];
#pragma unroll
  for (int t = 0; t < 4; ++t)
#pragma unroll
    for (int e = 0; e < 8; ++e) acc[t][e] = 0.f;

  async_copy_b128(ldsB[0], gsrc);      // prime buffer 0 (k-step 0)
  gsrc += 32;

  for (int kk = 0; kk < 16; ++kk) {
    const int k0 = kk * 32;
    if (kk < 15) {                      // stream next B tile
      async_copy_b128(ldsB[(kk + 1) & 1], gsrc);
      gsrc += 32;
    }
    // A tile (f32 -> bf16 convert) for this k-step
#pragma unroll
    for (int j = 0; j < 2; ++j) {
      const int ch = tid + j * 256;
      const int r = ch >> 2, c = (ch & 3) * 8;
      const float4* wp = (const float4*)&w[(size_t)(m0 + r) * CCH + k0 + c];
      *(v8bf*)&As[r][c] = cvt8(wp[0], wp[1]);
    }
    if (kk < 15) { ASYNC_WAIT_1; } else { ASYNC_WAIT_0; }
    __syncthreads();

    const v16bf a = load_frag(&As[wv * 16 + nn][0], hlf);
#pragma unroll
    for (int t = 0; t < 4; ++t) {
      const v16bf bb = load_frag(&Bs[kk & 1][t * 16 + nn][0], hlf);
      acc[t] = wmma_bf16(a, bb, acc[t]);
    }
    __syncthreads();
  }

  // Epilogue: bias, split q/k/v. q,k -> [b][h][l][d]; v -> [b][h][d][l].
#pragma unroll
  for (int t = 0; t < 4; ++t)
#pragma unroll
    for (int r = 0; r < 8; ++r) {
      const int o = m0 + wv * 16 + r + 8 * hlf;
      const int n = n0 + t * 16 + nn;
      const float val = acc[t][r] + bias[o];
      const int which = o >> 9;
      const int oo = o & 511;
      const int hd = oo >> 6, dd = oo & 63;
      if (which == 2) {
        vo[((size_t)(b * NHEADS + hd) * DHEAD + dd) * LSEQ + n] = (__bf16)val;
      } else {
        const size_t idx = ((size_t)(b * NHEADS + hd) * LSEQ + n) * DHEAD + dd;
        if (which == 0) qo[idx] = (__bf16)(val * 0.125f);
        else            ko[idx] = (__bf16)val;
      }
    }
}

// ---------------------------------------------------------------------------
// Kernel 3: causal flash attention; K/V tiles async double-buffered over jb.
// 128 threads (4 waves) own (b, head, 64 query rows).
// ---------------------------------------------------------------------------
__global__ __launch_bounds__(128) void attn_kernel(const __bf16* __restrict__ qg,
                                                   const __bf16* __restrict__ kg,
                                                   const __bf16* __restrict__ vg,
                                                   const float* __restrict__ ph,
                                                   const float* __restrict__ pw,
                                                   __bf16* __restrict__ og) {
  __shared__ __bf16 qs[64][80];          // [i][d]
  __shared__ __bf16 ks[2][64][80];       // [j][d]
  __shared__ __bf16 vs[2][64][80];       // [d][j]
  __shared__ __bf16 ps[4][16][80];       // per-wave P tile [m][j]

  const int tid  = threadIdx.x;
  const int lane = tid & 31;
  const int wv   = tid >> 5;
  const int hlf  = lane >> 4;
  const int nn   = lane & 15;
  const int h    = blockIdx.y;
  const int b    = blockIdx.z;
  const int i0   = blockIdx.x * 64;

  const __bf16* qh = qg + (size_t)(b * NHEADS + h) * LSEQ * DHEAD;  // [l][d]
  const __bf16* kh = kg + (size_t)(b * NHEADS + h) * LSEQ * DHEAD;  // [l][d]
  const __bf16* vh = vg + (size_t)(b * NHEADS + h) * DHEAD * LSEQ;  // [d][l]
  const float bias = ph[h] + pw[h];

  // stream K/V tiles for block jbn into buffer buf (8 async b128 per wave)
  auto issue_kv = [&](int jbn, int buf) {
    const int j0n = jbn * 64;
#pragma unroll
    for (int j = 0; j < 4; ++j) {
      const int ch = tid + j * 128;
      const int r = ch >> 3, c = (ch & 7) * 8;
      async_copy_b128(&ks[buf][r][c], kh + (size_t)(j0n + r) * DHEAD + c);
      async_copy_b128(&vs[buf][r][c], vh + (size_t)r * LSEQ + j0n + c);
    }
  };

  // q tile (async) + prime jb=0 K/V
#pragma unroll
  for (int j = 0; j < 4; ++j) {
    const int ch = tid + j * 128;
    const int r = ch >> 3, c = (ch & 7) * 8;
    async_copy_b128(&qs[r][c], qh + (size_t)(i0 + r) * DHEAD + c);
  }
  issue_kv(0, 0);

  float rmax[8], rsum[8];
  v8f acco[4];
#pragma unroll
  for (int r = 0; r < 8; ++r) { rmax[r] = -3.0e38f; rsum[r] = 0.f; }
#pragma unroll
  for (int t = 0; t < 4; ++t)
#pragma unroll
    for (int e = 0; e < 8; ++e) acco[t][e] = 0.f;

  v16bf aq0, aq1;
  bool have_q = false;

  for (int jb = 0; jb <= (int)blockIdx.x; ++jb) {
    const int j0 = jb * 64;
    const int buf = jb & 1;
    if (jb + 1 <= (int)blockIdx.x) {      // stream next K/V while computing
      issue_kv(jb + 1, buf ^ 1);
      ASYNC_WAIT_8;                       // current buffer's copies complete
    } else {
      ASYNC_WAIT_0;
    }
    __syncthreads();
    if (!have_q) {                        // q frags once, after first barrier
      aq0 = load_frag(&qs[wv * 16 + nn][0],  hlf);
      aq1 = load_frag(&qs[wv * 16 + nn][32], hlf);
      have_q = true;
    }

    // scores: S[i][j], K = d in two 32-chunks
    v8f sc[4];
#pragma unroll
    for (int t = 0; t < 4; ++t) {
      v8f s;
#pragma unroll
      for (int e = 0; e < 8; ++e) s[e] = 0.f;
      const v16bf b0 = load_frag(&ks[buf][t * 16 + nn][0],  hlf);
      const v16bf b1 = load_frag(&ks[buf][t * 16 + nn][32], hlf);
      s = wmma_bf16(aq0, b0, s);
      s = wmma_bf16(aq1, b1, s);
      const int jg = j0 + t * 16 + nn;
#pragma unroll
      for (int r = 0; r < 8; ++r) {
        const int ig = i0 + wv * 16 + r + 8 * hlf;
        float xv = s[r] + bias;
        if (jg > ig) xv = -3.0e38f;                    // causal mask
        s[r] = xv;
      }
      sc[t] = s;
    }

    // online softmax (rows live across the 16 lanes of each half-wave)
#pragma unroll
    for (int r = 0; r < 8; ++r) {
      float m = fmaxf(fmaxf(sc[0][r], sc[1][r]), fmaxf(sc[2][r], sc[3][r]));
#pragma unroll
      for (int off = 1; off < 16; off <<= 1) m = fmaxf(m, __shfl_xor(m, off, 32));
      const float newmax = fmaxf(rmax[r], m);
      const float scale  = __expf(rmax[r] - newmax);
      rmax[r] = newmax;
      float psum = 0.f;
#pragma unroll
      for (int t = 0; t < 4; ++t) {
        const float p = __expf(sc[t][r] - newmax);
        sc[t][r] = p;
        psum += p;
      }
#pragma unroll
      for (int off = 1; off < 16; off <<= 1) psum += __shfl_xor(psum, off, 32);
      rsum[r] = rsum[r] * scale + psum;
#pragma unroll
      for (int t = 0; t < 4; ++t) acco[t][r] *= scale;
    }

    // stage P (C-layout -> A-layout transpose) through per-wave LDS
#pragma unroll
    for (int t = 0; t < 4; ++t)
#pragma unroll
      for (int r = 0; r < 8; ++r)
        ps[wv][r + 8 * hlf][t * 16 + nn] = (__bf16)sc[t][r];
    asm volatile("s_wait_dscnt 0" ::: "memory");   // same-wave LDS RAW fence

    const v16bf ap0 = load_frag(&ps[wv][nn][0],  hlf);
    const v16bf ap1 = load_frag(&ps[wv][nn][32], hlf);
#pragma unroll
    for (int t = 0; t < 4; ++t) {
      const v16bf bv0 = load_frag(&vs[buf][t * 16 + nn][0],  hlf);
      const v16bf bv1 = load_frag(&vs[buf][t * 16 + nn][32], hlf);
      acco[t] = wmma_bf16(ap0, bv0, acco[t]);
      acco[t] = wmma_bf16(ap1, bv1, acco[t]);
    }
    __syncthreads();
  }

  // normalize; emit ao as [b][l][c]  (c = h*64+d contiguous for proj B tile)
#pragma unroll
  for (int r = 0; r < 8; ++r) {
    const int ig = i0 + wv * 16 + r + 8 * hlf;
    const float inv = 1.f / rsum[r];
#pragma unroll
    for (int t = 0; t < 4; ++t) {
      const int d = t * 16 + nn;
      og[((size_t)b * LSEQ + ig) * DMODEL + h * DHEAD + d] =
          (__bf16)(acco[t][r] * inv);
    }
  }
}

// ---------------------------------------------------------------------------
// Kernel 4: projection GEMM (512x512 @ 512xL per batch), f32 output + bias.
// Same async double-buffered B pipeline as the QKV GEMM.
// ---------------------------------------------------------------------------
__global__ __launch_bounds__(256) void proj_kernel(const __bf16* __restrict__ ao,
                                                   const float* __restrict__ w,
                                                   const float* __restrict__ bias,
                                                   float* __restrict__ out) {
  __shared__ __bf16 As[128][40];
  __shared__ __bf16 Bs[2][64][40];

  const int b    = blockIdx.z;
  const int m0   = blockIdx.y * 128;
  const int n0   = blockIdx.x * 64;
  const int tid  = threadIdx.x;
  const int lane = tid & 31;
  const int wv   = tid >> 5;
  const int hlf  = lane >> 4;
  const int nn   = lane & 15;

  const __bf16* xb = ao + (size_t)b * LSEQ * DMODEL;

  const int bn = tid >> 2, bc = (tid & 3) * 8;
  const __bf16* gsrc = xb + (size_t)(n0 + bn) * DMODEL + bc;
  void* ldsB[2] = { &Bs[0][bn][bc], &Bs[1][bn][bc] };

  v8f acc[4];
#pragma unroll
  for (int t = 0; t < 4; ++t)
#pragma unroll
    for (int e = 0; e < 8; ++e) acc[t][e] = 0.f;

  async_copy_b128(ldsB[0], gsrc);
  gsrc += 32;

  for (int kk = 0; kk < 16; ++kk) {
    const int k0 = kk * 32;
    if (kk < 15) {
      async_copy_b128(ldsB[(kk + 1) & 1], gsrc);
      gsrc += 32;
    }
#pragma unroll
    for (int j = 0; j < 2; ++j) {
      const int ch = tid + j * 256;
      const int r = ch >> 2, c = (ch & 3) * 8;
      const float4* wp = (const float4*)&w[(size_t)(m0 + r) * DMODEL + k0 + c];
      *(v8bf*)&As[r][c] = cvt8(wp[0], wp[1]);
    }
    if (kk < 15) { ASYNC_WAIT_1; } else { ASYNC_WAIT_0; }
    __syncthreads();

    const v16bf a = load_frag(&As[wv * 16 + nn][0], hlf);
#pragma unroll
    for (int t = 0; t < 4; ++t) {
      const v16bf bb = load_frag(&Bs[kk & 1][t * 16 + nn][0], hlf);
      acc[t] = wmma_bf16(a, bb, acc[t]);
    }
    __syncthreads();
  }

#pragma unroll
  for (int t = 0; t < 4; ++t)
#pragma unroll
    for (int r = 0; r < 8; ++r) {
      const int o = m0 + wv * 16 + r + 8 * hlf;
      const int n = n0 + t * 16 + nn;
      out[((size_t)b * DMODEL + o) * LSEQ + n] = acc[t][r] + bias[o];
    }
}

// ---------------------------------------------------------------------------
extern "C" void kernel_launch(void* const* d_in, const int* in_sizes, int n_in,
                              void* d_out, int out_size, void* d_ws,
                              size_t ws_size, hipStream_t stream) {
  const float* x      = (const float*)d_in[0];
  const float* ln_g   = (const float*)d_in[1];
  const float* ln_b   = (const float*)d_in[2];
  const float* qkv_w  = (const float*)d_in[3];
  const float* qkv_b  = (const float*)d_in[4];
  const float* proj_w = (const float*)d_in[5];
  const float* proj_b = (const float*)d_in[6];
  const float* pos_h  = (const float*)d_in[7];
  const float* pos_w  = (const float*)d_in[8];
  float* out = (float*)d_out;

  char* ws = (char*)d_ws;
  const size_t SEG = (size_t)BATCH * DMODEL * LSEQ * sizeof(__bf16);  // 8 MB
  __bf16* xn = (__bf16*)(ws);
  __bf16* q  = (__bf16*)(ws + 1 * SEG);
  __bf16* k  = (__bf16*)(ws + 2 * SEG);
  __bf16* v  = (__bf16*)(ws + 3 * SEG);
  __bf16* ao = (__bf16*)(ws + 4 * SEG);

  ln_kernel<<<dim3((BATCH * LSEQ) / 8), 256, 0, stream>>>(x, ln_g, ln_b, xn);
  qkv_kernel<<<dim3(LSEQ / 64, (3 * DMODEL) / 128, BATCH), 256, 0, stream>>>(
      xn, qkv_w, qkv_b, q, k, v);
  attn_kernel<<<dim3(LSEQ / 64, NHEADS, BATCH), 128, 0, stream>>>(
      q, k, v, pos_h, pos_w, ao);
  proj_kernel<<<dim3(LSEQ / 64, DMODEL / 128, BATCH), 256, 0, stream>>>(
      ao, proj_w, proj_b, out);
}
